// HPF_84894323572806
// MI455X (gfx1250) — matrix operand order repaired and checked by reading, not compiled, not run on hardware
//
#include <hip/hip_runtime.h>
#include <hip/hip_bf16.h>

typedef __attribute__((ext_vector_type(16))) _Float16 v16h;
typedef __attribute__((ext_vector_type(8)))  float    v8f;

#define SLOPE  0.05f
#define BN_EPS 1e-5f
#define HP     34          // padded plane edge (32 + 1 ring each side)
#define HP2    (34 * 34)   // padded plane elements

__device__ __forceinline__ v8f wmma16x16x32(v16h a, v16h b, v8f c) {
    return __builtin_amdgcn_wmma_f32_16x16x32_f16(false, a, false, b,
                                                  (short)0, c, false, false);
}

// ISA 16-bit A 16x32 layout: lanes0-15 K=0..7,16..23; lanes16-31 K=8..15,24..31
__device__ __forceinline__ int kmapA(int j, bool hi) {
    return (j & 7) + ((j >> 3) * 16) + (hi ? 8 : 0);
}
// ISA 16-bit B 32x16 layout: lanes0-15 K=0..7,16..23; lanes16-31 K=8..15,24..31
__device__ __forceinline__ int kmapB(int j, bool hi) {
    return (j & 7) + ((j >> 3) * 16) + (hi ? 8 : 0);
}

// ---------------------------------------------------------------------------
// First conv: 3x3 'SAME' + BN(eval) + LeakyReLU (scalar; 1.2 of 22 GFLOP).
// ---------------------------------------------------------------------------
__global__ void conv1_bn_leaky_kernel(const float* __restrict__ in,
                                      const float* __restrict__ w,
                                      const float* __restrict__ bias,
                                      const float* __restrict__ bn_g,
                                      const float* __restrict__ bn_b,
                                      const float* __restrict__ bn_m,
                                      const float* __restrict__ bn_v,
                                      float* __restrict__ out,
                                      int B, int C, int H, int W)
{
    int idx = blockIdx.x * blockDim.x + threadIdx.x;
    int total = B * C * H * W;
    if (idx >= total) return;
    int x = idx % W;
    int t = idx / W;
    int y = t % H;  t /= H;
    int co = t % C;
    int b  = t / C;

    const float* wbase = w + (size_t)co * C * 9;
    float acc = bias[co];
    for (int ci = 0; ci < C; ++ci) {
        const float* ip = in + ((size_t)(b * C + ci) * H) * W;
        const float* wp = wbase + ci * 9;
        #pragma unroll
        for (int ky = 0; ky < 3; ++ky) {
            int yy = y + ky - 1;
            if (yy < 0 || yy >= H) continue;
            #pragma unroll
            for (int kx = 0; kx < 3; ++kx) {
                int xx = x + kx - 1;
                if (xx < 0 || xx >= W) continue;
                acc += ip[yy * W + xx] * wp[ky * 3 + kx];
            }
        }
    }
    float inv = bn_g[co] * rsqrtf(bn_v[co] + BN_EPS);
    acc = (acc - bn_m[co]) * inv + bn_b[co];
    out[idx] = acc >= 0.f ? acc : SLOPE * acc;
}

// ---------------------------------------------------------------------------
// Zero an f16 buffer (establishes the zero pad rings deterministically).
// ---------------------------------------------------------------------------
__global__ void zero_f16_kernel(_Float16* __restrict__ p, int total)
{
    int i = blockIdx.x * blockDim.x + threadIdx.x;
    if (i < total) p[i] = (_Float16)0.f;
}

// ---------------------------------------------------------------------------
// NCHW f32 -> (B,N,C) f16   (GEMM/knn operand layout)
// ---------------------------------------------------------------------------
__global__ void pack_fh_kernel(const float* __restrict__ h,
                               _Float16* __restrict__ fh,
                               int B, int C, int N)
{
    int i = blockIdx.x * blockDim.x + threadIdx.x;
    if (i >= B * N * C) return;
    int c = i % C;
    int n = (i / C) % N;
    int b = i / (C * N);
    fh[i] = (_Float16)h[((size_t)b * C + c) * N + n];
}

// ---------------------------------------------------------------------------
// NCHW f32 -> zero-padded (B,C,34,34) f16 interior (ring pre-zeroed).
// ---------------------------------------------------------------------------
__global__ void pack_hpad_kernel(const float* __restrict__ h,
                                 _Float16* __restrict__ hp,
                                 int B, int C, int N)
{
    int i = blockIdx.x * blockDim.x + threadIdx.x;
    if (i >= B * C * N) return;
    int n = i % N;
    int c = (i / N) % C;
    int b = i / (C * N);
    int y = n >> 5, x = n & 31;
    hp[((size_t)(b * C + c) * HP + y + 1) * HP + x + 1] = (_Float16)h[i];
}

// ---------------------------------------------------------------------------
// Per-pixel squared norm (f32 for distance accuracy).
// ---------------------------------------------------------------------------
__global__ void sqnorm_kernel(const float* __restrict__ h,
                              float* __restrict__ sq,
                              int B, int C, int N)
{
    int i = blockIdx.x * blockDim.x + threadIdx.x;
    if (i >= B * N) return;
    int n = i % N;
    int b = i / N;
    float s = 0.f;
    for (int c = 0; c < C; ++c) {
        float v = h[((size_t)b * C + c) * N + n];
        s += v * v;
    }
    sq[i] = s;
}

// ---------------------------------------------------------------------------
// kNN graph: one wave per 16-row tile; Gram tile via WMMA (K = C = 32),
// d = sq[n] - 2G + sq[m] (+1e9 self), register insertion top-8 per row.
// ---------------------------------------------------------------------------
__global__ void __launch_bounds__(32)
knn_kernel(const _Float16* __restrict__ fh,
           const float* __restrict__ sq,
           int* __restrict__ graph,
           int B, int N, int C /* = 32 */)
{
    __shared__ float dtile[16][17];

    int mtiles = N / 16;
    int b  = blockIdx.x / mtiles;
    int mt = blockIdx.x % mtiles;
    int mbase = mt * 16;

    int lane = threadIdx.x & 31;
    int l15  = lane & 15;
    bool hi  = lane >= 16;

    const _Float16* fb = fh + (size_t)b * N * C;
    const float*    sb = sq + (size_t)b * N;

    v16h a;
    {
        const _Float16* ar = fb + (size_t)(mbase + l15) * C;
        #pragma unroll
        for (int j = 0; j < 16; ++j) a[j] = ar[kmapA(j, hi)];
    }

    float sqr8[8];
    #pragma unroll
    for (int r = 0; r < 8; ++r) sqr8[r] = sb[mbase + r + (hi ? 8 : 0)];

    float vals[8]; int idxs[8];
    #pragma unroll
    for (int k = 0; k < 8; ++k) { vals[k] = 3.0e38f; idxs[k] = 0; }

    for (int ct = 0; ct < N / 16; ++ct) {
        int cbase = ct * 16;
        v16h bf;
        const _Float16* br = fb + (size_t)(cbase + l15) * C;
        #pragma unroll
        for (int j = 0; j < 16; ++j) bf[j] = br[kmapB(j, hi)];

        v8f g = {};
        g = wmma16x16x32(a, bf, g);

        int col = cbase + l15;
        float sqc = sb[col];
        #pragma unroll
        for (int r = 0; r < 8; ++r) {
            int row = mbase + r + (hi ? 8 : 0);
            float dv = sqr8[r] - 2.0f * g[r] + sqc;
            if (row == col) dv += 1e9f;
            dtile[r + (hi ? 8 : 0)][l15] = dv;
        }
        __syncthreads();

        if (lane < 16) {
            for (int cc = 0; cc < 16; ++cc) {
                float v = dtile[lane][cc];
                int   id = cbase + cc;
                #pragma unroll
                for (int s = 0; s < 8; ++s) {
                    if (v < vals[s]) {
                        float tv = vals[s]; vals[s] = v; v = tv;
                        int   ti = idxs[s]; idxs[s] = id; id = ti;
                    }
                }
            }
        }
        __syncthreads();
    }

    if (lane < 16) {
        int row = mbase + lane;
        #pragma unroll
        for (int k = 0; k < 8; ++k)
            graph[((size_t)b * N + row) * 8 + k] = idxs[k];
    }
}

// ---------------------------------------------------------------------------
// meanNeigh - f, f16, layout (B,N,C).
// ---------------------------------------------------------------------------
__global__ void mdiff_kernel(const _Float16* __restrict__ fh,
                             const int* __restrict__ graph,
                             _Float16* __restrict__ md,
                             int B, int N, int C)
{
    int i = blockIdx.x * blockDim.x + threadIdx.x;
    if (i >= B * N * C) return;
    int c = i % C;
    int n = (i / C) % N;
    int b = i / (C * N);
    const int* g = graph + ((size_t)b * N + n) * 8;
    const _Float16* fb = fh + (size_t)b * N * C;
    float s = 0.f;
    #pragma unroll
    for (int k = 0; k < 8; ++k)
        s += (float)fb[(size_t)g[k] * C + c];
    md[i] = (_Float16)(s * 0.125f - (float)fb[(size_t)n * C + c]);
}

// ---------------------------------------------------------------------------
// Fully fused GConv layer, all-WMMA:
//   acc = im2col(hpad_in) x cw^T  (implicit GEMM, K = CIN*9)
//       + fh x wsw^T + md x dfw^T  (K = CIN each)
//   out = leaky(acc + cb + wsb + dfb)
// Epilogue writes next layer's operands directly (f16 padded NCHW + (B,N,C)),
// or f32 (B,COUT,N) for the final layer.
// One wave per 16x16 output tile.
// ---------------------------------------------------------------------------
template <int CIN, int COUT, bool LAST>
__global__ void __launch_bounds__(32)
gconv_fused_kernel(const _Float16* __restrict__ hpad_in,  // (B,CIN,34,34)
                   const _Float16* __restrict__ fh_in,    // (B,N,CIN)
                   const _Float16* __restrict__ md_in,    // (B,N,CIN)
                   const float* __restrict__ cw,          // (COUT,CIN,3,3)
                   const float* __restrict__ cb,
                   const float* __restrict__ wsw,         // (COUT,CIN)
                   const float* __restrict__ wsb,
                   const float* __restrict__ dfw,         // (COUT,CIN)
                   const float* __restrict__ dfb,
                   _Float16* __restrict__ hpad_out,       // (B,COUT,34,34)
                   _Float16* __restrict__ fh_out,         // (B,N,COUT)
                   float* __restrict__ out_f32,           // (B,COUT,N)
                   int B, int N)
{
    const int otiles = COUT / 16;
    const int mtiles = N / 16;
    int ot = blockIdx.x % otiles;
    int mt = (blockIdx.x / otiles) % mtiles;
    int b  = blockIdx.x / (otiles * mtiles);
    int mbase = mt * 16;
    int obase = ot * 16;

    int lane = threadIdx.x & 31;
    int l15  = lane & 15;
    bool hi  = lane >= 16;

    int m  = mbase + l15;          // this lane's A-row pixel
    int py = m >> 5, px = m & 31;  // unpadded coords; padded origin = (+1,+1),
                                   // conv tap (ky,kx) reads padded (py+ky, px+kx)
    int ocol = obase + l15;        // this lane's B-column (output channel)

    const _Float16* pb = hpad_in + (size_t)b * CIN * HP2;
    const _Float16* fb = fh_in   + (size_t)b * N * CIN;
    const _Float16* mb = md_in   + (size_t)b * N * CIN;
    const float*    wr = cw + (size_t)ocol * CIN * 9;

    v8f acc = {};

    // ---- conv branch: implicit-GEMM, K = CIN*9 (exact multiple of 32) ----
    #pragma unroll
    for (int t = 0; t < (CIN * 9) / 32; ++t) {
        v16h a, w;
        #pragma unroll
        for (int j = 0; j < 16; ++j) {
            int kk = t * 32 + kmapA(j, hi);     // compile-time constant
            int ci = kk / 9, rr = kk % 9;
            int ky = rr / 3, kx = rr % 3;
            a[j] = pb[(size_t)ci * HP2 + (py + ky) * HP + (px + kx)];
        }
        #pragma unroll
        for (int j = 0; j < 16; ++j)
            w[j] = (_Float16)wr[t * 32 + kmapB(j, hi)];
        acc = wmma16x16x32(a, w, acc);
    }

    // ---- non-local branch: f x wsw^T + (meanNeigh - f) x dfw^T ----
    #pragma unroll
    for (int k0 = 0; k0 < CIN; k0 += 32) {
        v16h a1, a2, b1, b2;
        const _Float16* ar1 = fb + (size_t)m * CIN + k0;
        const _Float16* ar2 = mb + (size_t)m * CIN + k0;
        #pragma unroll
        for (int j = 0; j < 16; ++j) {
            int kk = kmapA(j, hi);
            a1[j] = ar1[kk];
            a2[j] = ar2[kk];
        }
        const float* w1 = wsw + (size_t)ocol * CIN + k0;
        const float* w2 = dfw + (size_t)ocol * CIN + k0;
        #pragma unroll
        for (int j = 0; j < 16; ++j) {
            int kk = kmapB(j, hi);
            b1[j] = (_Float16)w1[kk];
            b2[j] = (_Float16)w2[kk];
        }
        acc = wmma16x16x32(a1, b1, acc);
        acc = wmma16x16x32(a2, b2, acc);
    }

    // ---- epilogue: bias + leaky, emit next layer's operand layouts ----
    float biasv = cb[ocol] + wsb[ocol] + dfb[ocol];
    #pragma unroll
    for (int r = 0; r < 8; ++r) {
        int pix = mbase + r + (hi ? 8 : 0);
        float v = acc[r] + biasv;
        v = v >= 0.f ? v : SLOPE * v;
        if constexpr (LAST) {
            out_f32[((size_t)b * COUT + ocol) * N + pix] = v;
        } else {
            _Float16 hv = (_Float16)v;
            fh_out[((size_t)b * N + pix) * COUT + ocol] = hv;
            int yy = pix >> 5, xx = pix & 31;
            hpad_out[((size_t)(b * COUT + ocol) * HP + yy + 1) * HP + xx + 1] = hv;
        }
    }
}

// ---------------------------------------------------------------------------
// Host launcher
// ---------------------------------------------------------------------------
extern "C" void kernel_launch(void* const* d_in, const int* in_sizes, int n_in,
                              void* d_out, int out_size, void* d_ws, size_t ws_size,
                              hipStream_t stream)
{
    (void)in_sizes; (void)n_in; (void)out_size; (void)ws_size;

    const int B = 64, H = 32, W = 32, N = H * W;
    const int IC = 32, OC = 64;

    const float* x      = (const float*)d_in[0];
    const float* conv_w = (const float*)d_in[1];
    const float* conv_b = (const float*)d_in[2];
    const float* bn_g   = (const float*)d_in[3];
    const float* bn_b   = (const float*)d_in[4];
    const float* bn_m   = (const float*)d_in[5];
    const float* bn_v   = (const float*)d_in[6];

    // workspace carve-up (bytes)
    char* ws = (char*)d_ws;
    float*    h1    = (float*)   (ws +  (size_t)0);                     // 8 MB  (B,32,N) f32
    _Float16* fhA   = (_Float16*)(ws + ((size_t)8  << 20));             // 8 MB  (B,N,<=64) f16
    _Float16* fhB   = (_Float16*)(ws + ((size_t)16 << 20));             // 8 MB
    _Float16* md    = (_Float16*)(ws + ((size_t)24 << 20));             // 8 MB
    _Float16* hpadA = (_Float16*)(ws + ((size_t)32 << 20));             // 10 MB (B,<=64,34,34) f16
    _Float16* hpadB = (_Float16*)(ws + ((size_t)42 << 20));             // 10 MB
    float*    sq    = (float*)   (ws + ((size_t)52 << 20));             // 256 KB
    int*      graph = (int*)     (ws + ((size_t)52 << 20) + ((size_t)1 << 18)); // 2 MB

    const int TB = 256;
    auto cdiv = [](int a, int b) { return (a + b - 1) / b; };
    const int hpad_elems = B * OC * HP2;

    // 0) establish zero pad rings (interiors get overwritten every call)
    zero_f16_kernel<<<cdiv(hpad_elems, TB), TB, 0, stream>>>(hpadA, hpad_elems);
    zero_f16_kernel<<<cdiv(hpad_elems, TB), TB, 0, stream>>>(hpadB, hpad_elems);

    // 1) conv + BN + leaky  ->  h1 (B,32,H,W) f32
    conv1_bn_leaky_kernel<<<cdiv(B * IC * N, TB), TB, 0, stream>>>(
        x, conv_w, conv_b, bn_g, bn_b, bn_m, bn_v, h1, B, IC, H, W);

    // 2) operand packing for knn + gc1
    pack_fh_kernel  <<<cdiv(B * N * IC, TB), TB, 0, stream>>>(h1, fhA, B, IC, N);
    pack_hpad_kernel<<<cdiv(B * IC * N, TB), TB, 0, stream>>>(h1, hpadA, B, IC, N);
    sqnorm_kernel   <<<cdiv(B * N,      TB), TB, 0, stream>>>(h1, sq, B, IC, N);

    // 3) kNN graph (shared by all three gconvs)
    knn_kernel<<<B * (N / 16), 32, 0, stream>>>(fhA, sq, graph, B, N, IC);

    const float** p = (const float**)d_in;

    // 4) gc1: 32 -> 32   (in: hpadA/fhA, out: hpadB/fhB)
    mdiff_kernel<<<cdiv(B * N * IC, TB), TB, 0, stream>>>(fhA, graph, md, B, N, IC);
    gconv_fused_kernel<32, 32, false><<<B * (N / 16) * (32 / 16), 32, 0, stream>>>(
        hpadA, fhA, md, p[7], p[8], p[9], p[10], p[11], p[12],
        hpadB, fhB, nullptr, B, N);

    // 5) gc2: 32 -> 64   (in: hpadB/fhB, out: hpadA/fhA)
    mdiff_kernel<<<cdiv(B * N * IC, TB), TB, 0, stream>>>(fhB, graph, md, B, N, IC);
    gconv_fused_kernel<32, 64, false><<<B * (N / 16) * (64 / 16), 32, 0, stream>>>(
        hpadB, fhB, md, p[13], p[14], p[15], p[16], p[17], p[18],
        hpadA, fhA, nullptr, B, N);

    // 6) gc3: 64 -> 64   (in: hpadA/fhA, out: d_out f32)
    mdiff_kernel<<<cdiv(B * N * OC, TB), TB, 0, stream>>>(fhA, graph, md, B, N, OC);
    gconv_fused_kernel<64, 64, true><<<B * (N / 16) * (64 / 16), 32, 0, stream>>>(
        hpadA, fhA, md, p[19], p[20], p[21], p[22], p[23], p[24],
        nullptr, nullptr, (float*)d_out, B, N);
}
